// NodeEmbedder_14525579395502
// MI455X (gfx1250) — matrix operand order repaired and checked by reading, not compiled, not run on hardware
//
#include <hip/hip_runtime.h>
#include <hip/hip_bf16.h>
#include <float.h>

// Problem constants (match the reference)
#define NN       50000
#define EE       800000
#define IN_DIMC  300
#define KPAD_IN  320
#define HIDC     64
#define HCC      128
#define EDGE_D   16
#define NHEAD    2
#define CHAN     64
#define NLAYERS  2

typedef __bf16 bf16_t;
typedef __attribute__((ext_vector_type(8)))  __bf16 bf16x8;
typedef __attribute__((ext_vector_type(16))) __bf16 bf16x16;
typedef __attribute__((ext_vector_type(8)))  float  f32x8;

__device__ __forceinline__ void atomicMaxF(float* addr, float v) {
    // Monotonic int/uint ordering trick; mx is initialized to -FLT_MAX.
    if (v >= 0.0f) atomicMax((int*)addr, __float_as_int(v));
    else           atomicMin((unsigned int*)addr, __float_as_uint(v));
}

// ---------------------------------------------------------------------------
// Generic bf16 WMMA GEMM: C[M x Nout] = A[M x K] * B[K x Nout] + bias
// A: bf16 row-major (lda); BT: B transposed, bf16 row-major Nout x K (ldb=K).
// Each wave (wave32) computes one 16x16 tile with v_wmma_f32_16x16x32_bf16.
// ---------------------------------------------------------------------------
__global__ __launch_bounds__(256)
void gemm_bf16_wmma(const bf16_t* __restrict__ A, int lda,
                    const bf16_t* __restrict__ BT, int ldb,
                    const float* __restrict__ bias,
                    float* __restrict__ Cf, bf16_t* __restrict__ Cb, int ldc,
                    int M, int Nout, int K, int do_relu)
{
    const int wave = threadIdx.x >> 5;
    const int lane = threadIdx.x & 31;
    const int tiles_n = Nout >> 4;
    const int tiles_total = (M >> 4) * tiles_n;
    const int tile = blockIdx.x * 8 + wave;
    if (tile >= tiles_total) return;           // wave-uniform; EXEC stays full
    const int tm = tile / tiles_n;
    const int tn = tile % tiles_n;
    const int sub  = lane & 15;
    const int half = lane >> 4;

    // A layout (16-bit 16x32): lane holds row M=sub; K chunks [8*half,8*half+8)
    // and [16+8*half, 16+8*half+8).
    const bf16_t* arow = A  + (size_t)(tm * 16 + sub) * lda;
    // B layout (16-bit 32x16): lane holds col N=sub; K chunks [8*half,8*half+8)
    // and [16+8*half, 16+8*half+8) (BT stored as Nout x K row-major).
    const bf16_t* brow = BT + (size_t)(tn * 16 + sub) * ldb;

    f32x8 acc = {0.f, 0.f, 0.f, 0.f, 0.f, 0.f, 0.f, 0.f};
    for (int k0 = 0; k0 < K; k0 += 32) {
        bf16x8 a0 = *(const bf16x8*)(arow + k0 + 8 * half);
        bf16x8 a1 = *(const bf16x8*)(arow + k0 + 16 + 8 * half);
        bf16x16 a;
#pragma unroll
        for (int i = 0; i < 8; ++i) { a[i] = a0[i]; a[i + 8] = a1[i]; }
        bf16x8 b0 = *(const bf16x8*)(brow + k0 + 8 * half);
        bf16x8 b1 = *(const bf16x8*)(brow + k0 + 16 + 8 * half);
        bf16x16 b;
#pragma unroll
        for (int i = 0; i < 8; ++i) { b[i] = b0[i]; b[i + 8] = b1[i]; }
        acc = __builtin_amdgcn_wmma_f32_16x16x32_bf16(
            /*neg_a=*/false, a, /*neg_b=*/false, b,
            /*c_mod=*/(short)0, acc, /*reuse_a=*/false, /*reuse_b=*/false);
    }

    // C/D layout: col = tn*16 + sub ; row = tm*16 + r + 8*half for VGPR r.
    const int col = tn * 16 + sub;
    const float bv = bias ? bias[col] : 0.f;
#pragma unroll
    for (int r = 0; r < 8; ++r) {
        const int m = tm * 16 + r + 8 * half;
        float v = acc[r] + bv;
        if (do_relu) v = v > 0.f ? v : 0.f;
        if (Cf) Cf[(size_t)m * ldc + col] = v;
        if (Cb) Cb[(size_t)m * ldc + col] = (bf16_t)v;
    }
}

// ---------------------------------------------------------------------------
// BatchNorm statistics: column sums / sums-of-squares. 200 blocks x 250 rows.
// ---------------------------------------------------------------------------
__global__ __launch_bounds__(256)
void bn_stats_kernel(const float* __restrict__ x, float* gsum, float* gsq)
{
    const int tid = threadIdx.x;
    const int r0 = blockIdx.x * 250;
    const int c0 = tid, c1 = tid + 256;
    float s0 = 0.f, q0 = 0.f, s1 = 0.f, q1 = 0.f;
    for (int i = 0; i < 250; ++i) {
        const float* row = x + (size_t)(r0 + i) * IN_DIMC;
        float v = row[c0]; s0 += v; q0 += v * v;
        if (c1 < IN_DIMC) { float w = row[c1]; s1 += w; q1 += w * w; }
    }
    atomicAdd(&gsum[c0], s0); atomicAdd(&gsq[c0], q0);
    if (c1 < IN_DIMC) { atomicAdd(&gsum[c1], s1); atomicAdd(&gsq[c1], q1); }
}

__global__ void bn_finalize_kernel(const float* gsum, const float* gsq,
                                   float* mean, float* rstd)
{
    int c = blockIdx.x * blockDim.x + threadIdx.x;
    if (c >= IN_DIMC) return;
    float m = gsum[c] * (1.0f / NN);
    float v = gsq[c] * (1.0f / NN) - m * m;
    mean[c] = m;
    rstd[c] = rsqrtf(v + 1e-5f);
}

// Normalize + scale/shift, write bf16, zero-pad K to 320.
__global__ __launch_bounds__(256)
void bn_apply_kernel(const float* __restrict__ x, const float* __restrict__ mean,
                     const float* __restrict__ rstd, const float* __restrict__ bnw,
                     const float* __restrict__ bnb, bf16_t* __restrict__ xn)
{
    int idx = blockIdx.x * 256 + threadIdx.x;
    if (idx >= NN * KPAD_IN) return;
    int c = idx % KPAD_IN;
    float v = 0.f;
    if (c < IN_DIMC) {
        int n = idx / KPAD_IN;
        v = (x[(size_t)n * IN_DIMC + c] - mean[c]) * rstd[c] * bnw[c] + bnb[c];
    }
    xn[idx] = (bf16_t)v;
}

// Transpose + convert weight W[K x Nout] (fp32) -> WT[Nout x Kpad] (bf16, 0-pad)
__global__ void prep_wT_kernel(const float* __restrict__ W, bf16_t* __restrict__ WT,
                               int K, int Kpad, int Nout)
{
    int idx = blockIdx.x * blockDim.x + threadIdx.x;
    if (idx >= Nout * Kpad) return;
    int n = idx / Kpad, k = idx % Kpad;
    WT[idx] = (k < K) ? (bf16_t)W[(size_t)k * Nout + n] : (bf16_t)0.f;
}

__global__ void zero_f32_kernel(float* p, int n)
{
    int idx = blockIdx.x * blockDim.x + threadIdx.x;
    if (idx < n) p[idx] = 0.f;
}

// Reset per-layer edge accumulators: agg=0 (N*HC), mx=-FLT_MAX, denom=0 (N*H)
__global__ __launch_bounds__(256)
void layer_reset_kernel(float* mx, float* denom, float* agg)
{
    int idx = blockIdx.x * 256 + threadIdx.x;
    if (idx >= NN * HCC) return;
    agg[idx] = 0.f;
    if (idx < NN * NHEAD) { mx[idx] = -FLT_MAX; denom[idx] = 0.f; }
}

// ---------------------------------------------------------------------------
// Edge pass 1: logits[e,h] = att_h . leaky_relu(xl[src]+xr[dst]+edge_attr@We)
// One wave per (edge, head); We/att cached in LDS; atomic segment max.
// ---------------------------------------------------------------------------
__global__ __launch_bounds__(256)
void edge_logits_kernel(const int* __restrict__ src, const int* __restrict__ dst,
                        const float* __restrict__ edge_attr,
                        const float* __restrict__ We, const float* __restrict__ attw,
                        const float* __restrict__ xl, const float* __restrict__ xr,
                        float* __restrict__ logits, float* __restrict__ mx)
{
    __shared__ float sWe[EDGE_D * HCC];
    __shared__ float sAtt[HCC];
    for (int i = threadIdx.x; i < EDGE_D * HCC; i += 256) sWe[i] = We[i];
    if (threadIdx.x < HCC) sAtt[threadIdx.x] = attw[threadIdx.x];
    __syncthreads();

    const int wid = blockIdx.x * 8 + (threadIdx.x >> 5);
    if (wid >= EE * NHEAD) return;
    const int lane = threadIdx.x & 31;
    const int e = wid >> 1, hd = wid & 1;
    const int s = src[e], d = dst[e];

    float ea[EDGE_D];
    const float* eap = edge_attr + (size_t)e * EDGE_D;
#pragma unroll
    for (int k = 0; k < EDGE_D; ++k) ea[k] = eap[k];

    float acc = 0.f;
#pragma unroll
    for (int j = 0; j < 2; ++j) {
        const int c = hd * CHAN + lane + j * 32;
        float eec = 0.f;
#pragma unroll
        for (int k = 0; k < EDGE_D; ++k) eec += ea[k] * sWe[k * HCC + c];
        float v = xl[(size_t)s * HCC + c] + xr[(size_t)d * HCC + c] + eec;
        v = v > 0.f ? v : 0.2f * v;                 // leaky_relu(0.2)
        acc += v * sAtt[c];
    }
#pragma unroll
    for (int off = 16; off; off >>= 1) acc += __shfl_xor(acc, off, 32);
    if (lane == 0) {
        logits[wid] = acc;
        atomicMaxF(&mx[d * NHEAD + hd], acc);
    }
}

// Edge pass 2: p = exp(logit - mx[dst]); denom[dst] += p   (one thread each)
__global__ __launch_bounds__(256)
void edge_softmax_kernel(const int* __restrict__ dst, float* __restrict__ logits,
                         const float* __restrict__ mx, float* __restrict__ denom)
{
    int idx = blockIdx.x * 256 + threadIdx.x;
    if (idx >= EE * NHEAD) return;
    const int e = idx >> 1, hd = idx & 1;
    const int d = dst[e];
    float p = __expf(logits[idx] - mx[d * NHEAD + hd]);
    logits[idx] = p;                                // overwrite with numerator
    atomicAdd(&denom[d * NHEAD + hd], p);
}

// Edge pass 3: agg[dst] += alpha * xl[src]   (one wave per (edge, head))
__global__ __launch_bounds__(256)
void edge_aggregate_kernel(const int* __restrict__ src, const int* __restrict__ dst,
                           const float* __restrict__ p, const float* __restrict__ denom,
                           const float* __restrict__ xl, float* __restrict__ agg)
{
    const int wid = blockIdx.x * 8 + (threadIdx.x >> 5);
    if (wid >= EE * NHEAD) return;
    const int lane = threadIdx.x & 31;
    const int e = wid >> 1, hd = wid & 1;
    const int s = src[e], d = dst[e];
    const float alpha = p[wid] / denom[d * NHEAD + hd];
#pragma unroll
    for (int j = 0; j < 2; ++j) {
        const int c = hd * CHAN + lane + j * 32;
        atomicAdd(&agg[(size_t)d * HCC + c], xl[(size_t)s * HCC + c] * alpha);
    }
}

// relu(agg + conv_bias) -> bf16 (feeds the per-layer lin GEMM)
__global__ __launch_bounds__(256)
void relu_bias_bf16_kernel(const float* __restrict__ agg, const float* __restrict__ cbias,
                           bf16_t* __restrict__ out)
{
    int idx = blockIdx.x * 256 + threadIdx.x;
    if (idx >= NN * HCC) return;
    float v = agg[idx] + cbias[idx & (HCC - 1)];
    out[idx] = (bf16_t)(v > 0.f ? v : 0.f);
}

// ---------------------------------------------------------------------------
extern "C" void kernel_launch(void* const* d_in, const int* in_sizes, int n_in,
                              void* d_out, int out_size, void* d_ws, size_t ws_size,
                              hipStream_t stream)
{
    (void)in_sizes; (void)n_in; (void)out_size; (void)ws_size;

    const float* x          = (const float*)d_in[0];
    const int*   edge_index = (const int*)d_in[1];
    const float* edge_attr  = (const float*)d_in[2];
    const float* bn_w       = (const float*)d_in[3];
    const float* bn_b       = (const float*)d_in[4];
    const float* proj_W     = (const float*)d_in[5];
    const float* proj_b     = (const float*)d_in[6];
    const float* lin_l_W    = (const float*)d_in[7];
    const float* lin_l_b    = (const float*)d_in[8];
    const float* lin_r_W    = (const float*)d_in[9];
    const float* lin_r_b    = (const float*)d_in[10];
    const float* lin_edge_W = (const float*)d_in[11];
    const float* att        = (const float*)d_in[12];
    const float* conv_bias  = (const float*)d_in[13];
    const float* lin_W      = (const float*)d_in[14];
    const float* lin_b      = (const float*)d_in[15];
    const float* out_W      = (const float*)d_in[16];
    const float* out_b      = (const float*)d_in[17];

    const int* src = edge_index;
    const int* dst = edge_index + EE;

    // Workspace bump allocator (256B aligned).
    size_t off = 0;
    auto alloc = [&](size_t bytes) -> char* {
        char* p = (char*)d_ws + off;
        off += (bytes + 255) & ~(size_t)255;
        return p;
    };
    float*  gsum   = (float*)alloc(IN_DIMC * 4);
    float*  gsq    = (float*)alloc(IN_DIMC * 4);
    float*  meanp  = (float*)alloc(IN_DIMC * 4);
    float*  rstdp  = (float*)alloc(IN_DIMC * 4);
    bf16_t* xn     = (bf16_t*)alloc((size_t)NN * KPAD_IN * 2);
    bf16_t* h_bf   = (bf16_t*)alloc((size_t)NN * HIDC * 2);
    bf16_t* hrelu  = (bf16_t*)alloc((size_t)NN * HCC * 2);
    float*  xl     = (float*)alloc((size_t)NN * HCC * 4);
    float*  xr     = (float*)alloc((size_t)NN * HCC * 4);
    float*  agg    = (float*)alloc((size_t)NN * HCC * 4);
    float*  logits = (float*)alloc((size_t)EE * NHEAD * 4);
    float*  mx     = (float*)alloc((size_t)NN * NHEAD * 4);
    float*  denom  = (float*)alloc((size_t)NN * NHEAD * 4);
    bf16_t* projT  = (bf16_t*)alloc((size_t)HIDC * KPAD_IN * 2);
    bf16_t* WlT0   = (bf16_t*)alloc((size_t)HCC * HIDC * 2);
    bf16_t* WlT1   = (bf16_t*)alloc((size_t)HCC * HIDC * 2);
    bf16_t* WrT0   = (bf16_t*)alloc((size_t)HCC * HIDC * 2);
    bf16_t* WrT1   = (bf16_t*)alloc((size_t)HCC * HIDC * 2);
    bf16_t* linT0  = (bf16_t*)alloc((size_t)HIDC * HCC * 2);
    bf16_t* linT1  = (bf16_t*)alloc((size_t)HIDC * HCC * 2);
    bf16_t* outT   = (bf16_t*)alloc((size_t)HIDC * HIDC * 2);
    bf16_t* WlT[2] = {WlT0, WlT1};
    bf16_t* WrT[2] = {WrT0, WrT1};
    bf16_t* linT[2] = {linT0, linT1};

    // --- BatchNorm statistics ---
    zero_f32_kernel<<<2, 256, 0, stream>>>(gsum, IN_DIMC);
    zero_f32_kernel<<<2, 256, 0, stream>>>(gsq, IN_DIMC);
    bn_stats_kernel<<<200, 256, 0, stream>>>(x, gsum, gsq);
    bn_finalize_kernel<<<2, 256, 0, stream>>>(gsum, gsq, meanp, rstdp);
    bn_apply_kernel<<<(NN * KPAD_IN + 255) / 256, 256, 0, stream>>>(
        x, meanp, rstdp, bn_w, bn_b, xn);

    // --- Weight conversion / transpose / pad (bf16) ---
    prep_wT_kernel<<<(HIDC * KPAD_IN + 255) / 256, 256, 0, stream>>>(
        proj_W, projT, IN_DIMC, KPAD_IN, HIDC);
    for (int l = 0; l < NLAYERS; ++l) {
        prep_wT_kernel<<<(HCC * HIDC + 255) / 256, 256, 0, stream>>>(
            lin_l_W + (size_t)l * HIDC * HCC, WlT[l], HIDC, HIDC, HCC);
        prep_wT_kernel<<<(HCC * HIDC + 255) / 256, 256, 0, stream>>>(
            lin_r_W + (size_t)l * HIDC * HCC, WrT[l], HIDC, HIDC, HCC);
        prep_wT_kernel<<<(HIDC * HCC + 255) / 256, 256, 0, stream>>>(
            lin_W + (size_t)l * HCC * HIDC, linT[l], HCC, HCC, HIDC);
    }
    prep_wT_kernel<<<(HIDC * HIDC + 255) / 256, 256, 0, stream>>>(
        out_W, outT, HIDC, HIDC, HIDC);

    const int tiles_m = NN / 16;  // 3125
    // --- Projection GEMM: h = bn(x) @ proj_W + proj_b  (bf16 out) ---
    {
        int tiles = tiles_m * (HIDC / 16);
        gemm_bf16_wmma<<<(tiles + 7) / 8, 256, 0, stream>>>(
            xn, KPAD_IN, projT, KPAD_IN, proj_b,
            nullptr, h_bf, HIDC, NN, HIDC, KPAD_IN, 0);
    }

    // --- GATv2 layers ---
    for (int l = 0; l < NLAYERS; ++l) {
        int tiles128 = tiles_m * (HCC / 16);
        gemm_bf16_wmma<<<(tiles128 + 7) / 8, 256, 0, stream>>>(
            h_bf, HIDC, WlT[l], HIDC, lin_l_b + l * HCC,
            xl, nullptr, HCC, NN, HCC, HIDC, 0);
        gemm_bf16_wmma<<<(tiles128 + 7) / 8, 256, 0, stream>>>(
            h_bf, HIDC, WrT[l], HIDC, lin_r_b + l * HCC,
            xr, nullptr, HCC, NN, HCC, HIDC, 0);

        layer_reset_kernel<<<(NN * HCC + 255) / 256, 256, 0, stream>>>(mx, denom, agg);

        const int ewaves = EE * NHEAD;               // 1.6M waves
        edge_logits_kernel<<<(ewaves + 7) / 8, 256, 0, stream>>>(
            src, dst, edge_attr,
            lin_edge_W + (size_t)l * EDGE_D * HCC, att + (size_t)l * HCC,
            xl, xr, logits, mx);
        edge_softmax_kernel<<<(ewaves + 255) / 256, 256, 0, stream>>>(
            dst, logits, mx, denom);
        edge_aggregate_kernel<<<(ewaves + 7) / 8, 256, 0, stream>>>(
            src, dst, logits, denom, xl, agg);

        relu_bias_bf16_kernel<<<(NN * HCC + 255) / 256, 256, 0, stream>>>(
            agg, conv_bias + l * HCC, hrelu);

        int tiles64 = tiles_m * (HIDC / 16);
        gemm_bf16_wmma<<<(tiles64 + 7) / 8, 256, 0, stream>>>(
            hrelu, HCC, linT[l], HCC, lin_b + l * HIDC,
            nullptr, h_bf, HIDC, NN, HIDC, HCC, 1);
    }

    // --- Output GEMM: out = h @ out_W + out_b (fp32) ---
    {
        int tiles = tiles_m * (HIDC / 16);
        gemm_bf16_wmma<<<(tiles + 7) / 8, 256, 0, stream>>>(
            h_bf, HIDC, outT, HIDC, out_b,
            (float*)d_out, nullptr, HIDC, NN, HIDC, HIDC, 0);
    }
}
